// TotalLoss_28595892256874
// MI455X (gfx1250) — matrix operand order repaired and checked by reading, not compiled, not run on hardware
//
#include <hip/hip_runtime.h>
#include <hip/hip_bf16.h>

typedef __attribute__((ext_vector_type(16))) _Float16     v16h;
typedef __attribute__((ext_vector_type(8)))  float        v8f;
typedef __attribute__((ext_vector_type(4)))  unsigned int u32x4;
typedef __attribute__((ext_vector_type(2)))  unsigned int u32x2;
typedef __attribute__((ext_vector_type(4)))  float        f32x4;
typedef __attribute__((ext_vector_type(4)))  int          i32x4;

#define N_ROWS      8192
#define DIM         1024
#define NUM_CLASSES 200
#define NUM_BATCHES 128

#define BLK         256      // I-block and J-block size (256x256 tile per workgroup)
#define KBLK        32       // K slice per LDS stage (matches WMMA K=32)
#define LDS_STRIDE  40       // halves per LDS row: 32 data + 8 pad (80B, conflict-free b128)
#define ISPLIT      16       // row range split across gridDim.y
#define NTHREADS    512      // 16 waves: 4x4 wave grid, each wave owns 64x64

// ---------------- kernel 0: zero row accumulators ----------------
__global__ void zero_acc_kernel(float* __restrict__ acc, int n) {
    int t = blockIdx.x * blockDim.x + threadIdx.x;
    if (t < n) acc[t] = 0.0f;
}

// ---------------- kernel 1: row normalize f32 -> f16 ----------------
__global__ __launch_bounds__(256)
void normalize_rows_kernel(const float* __restrict__ z, _Float16* __restrict__ zn) {
    const int row = blockIdx.x;
    const int t   = threadIdx.x;           // 256 threads, 4 floats each
    const f32x4* zr = (const f32x4*)(z + (size_t)row * DIM);
    f32x4 v = zr[t];
    float s = v.x * v.x + v.y * v.y + v.z * v.z + v.w * v.w;
    #pragma unroll
    for (int off = 16; off >= 1; off >>= 1) s += __shfl_xor(s, off, 32);
    __shared__ float wsum[8];
    const int wave = t >> 5, lane = t & 31;
    if (lane == 0) wsum[wave] = s;
    __syncthreads();
    float tot = wsum[0] + wsum[1] + wsum[2] + wsum[3] +
                wsum[4] + wsum[5] + wsum[6] + wsum[7];
    const float inv = 1.0f / fmaxf(sqrtf(tot), 1e-8f);
    union { _Float16 h[4]; u32x2 u; } pk;
    pk.h[0] = (_Float16)(v.x * inv);
    pk.h[1] = (_Float16)(v.y * inv);
    pk.h[2] = (_Float16)(v.z * inv);
    pk.h[3] = (_Float16)(v.w * inv);
    ((u32x2*)(zn + (size_t)row * DIM))[t] = pk.u;
}

// ---------------- kernel 2: fused Gram + exp + masked column sums ----------------
// S = Zn*Zn^T is symmetric with symmetric masks, so masked row sums == masked
// column sums. Each workgroup owns a fixed 256-column block (j0) and loops over
// a 512-row range; per-lane column accumulators live in registers for the whole
// loop (no shuffles, no per-block atomics). 16 waves in 4x4 grid; each wave
// computes a 64x64 subtile = 4x4 WMMA tiles of 16x16.
__global__ __launch_bounds__(NTHREADS)
void gram_epilogue_kernel(const _Float16* __restrict__ zn,
                          const int* __restrict__ labels,
                          float* __restrict__ row_same,
                          float* __restrict__ row_diff) {
    __shared__ __align__(16) _Float16 lA[BLK * LDS_STRIDE];   // rows (i side, streamed)
    __shared__ __align__(16) _Float16 lB[BLK * LDS_STRIDE];   // cols (j side)
    __shared__ __align__(16) int   labI[BLK];
    __shared__ __align__(16) int   labJ[BLK];
    __shared__ float accSl[BLK];
    __shared__ float accDl[BLK];

    const int t    = threadIdx.x;
    const int lane = t & 31;
    const int wave = t >> 5;          // 0..15
    const int wm   = wave >> 2;       // 0..3 : 64-row band within block
    const int wn   = wave & 3;        // 0..3 : 64-col band within block
    const int hh   = lane >> 4;       // half-wave 0/1
    const int l16  = lane & 15;

    const int j0 = blockIdx.x * BLK;                    // fixed column block
    const int ibeg = blockIdx.y * (N_ROWS / ISPLIT);    // 512-row range
    const int iend = ibeg + (N_ROWS / ISPLIT);

    if (t < BLK) {
        labJ[t]  = labels[j0 + t];
        accSl[t] = 0.0f;
        accDl[t] = 0.0f;
    }
    __syncthreads();

    // per-lane fixed columns: jcol(n) = wn*64 + n*16 + l16
    int lj[4];
    int jg[4];
    #pragma unroll
    for (int n = 0; n < 4; ++n) {
        const int jc = wn * 64 + n * 16 + l16;
        lj[n] = labJ[jc];
        jg[n] = j0 + jc;
    }
    float cS[4] = {0.f, 0.f, 0.f, 0.f};
    float cD[4] = {0.f, 0.f, 0.f, 0.f};

    for (int ib0 = ibeg; ib0 < iend; ib0 += BLK) {
        __syncthreads();                       // prior epilogue done with labI
        if (t < BLK) labI[t] = labels[ib0 + t];

        v8f c[4][4];
        #pragma unroll
        for (int m = 0; m < 4; ++m)
            #pragma unroll
            for (int n = 0; n < 4; ++n)
                c[m][n] = (v8f){0.f,0.f,0.f,0.f,0.f,0.f,0.f,0.f};

        for (int k0 = 0; k0 < DIM; k0 += KBLK) {
            __syncthreads();                   // previous slice consumed
            // stage 256 rows x 32 halves for each of A (i rows) and B (j rows)
            #pragma unroll
            for (int p = 0; p < 2; ++p) {
                const int idx = t + p * NTHREADS;   // 0..1023
                const int row = idx >> 2;
                const int cp  = idx & 3;            // 16B chunk in row
                *(u32x4*)(lA + row * LDS_STRIDE + cp * 8) =
                    *(const u32x4*)(zn + (size_t)(ib0 + row) * DIM + k0 + cp * 8);
                *(u32x4*)(lB + row * LDS_STRIDE + cp * 8) =
                    *(const u32x4*)(zn + (size_t)(j0 + row) * DIM + k0 + cp * 8);
            }
            __syncthreads();

            // A fragments: lane<16 -> K 0..7 & 16..23 ; lane>=16 -> K 8..15 & 24..31
            v16h a[4];
            #pragma unroll
            for (int m = 0; m < 4; ++m) {
                const int r = wm * 64 + m * 16 + l16;
                union { u32x4 u[2]; v16h h; } ua;
                ua.u[0] = *(const u32x4*)(lA + r * LDS_STRIDE + hh * 8);
                ua.u[1] = *(const u32x4*)(lA + r * LDS_STRIDE + hh * 8 + 16);
                a[m] = ua.h;
            }
            // B fragments: lane<16 -> K 0..7 & 16..23 ; lane>=16 -> K 8..15 & 24..31
            v16h b[4];
            #pragma unroll
            for (int n = 0; n < 4; ++n) {
                const int r = wn * 64 + n * 16 + l16;
                union { u32x4 u[2]; v16h h; } ub;
                ub.u[0] = *(const u32x4*)(lB + r * LDS_STRIDE + hh * 8);
                ub.u[1] = *(const u32x4*)(lB + r * LDS_STRIDE + hh * 8 + 16);
                b[n] = ub.h;
            }
            #pragma unroll
            for (int m = 0; m < 4; ++m)
                #pragma unroll
                for (int n = 0; n < 4; ++n)
                    c[m][n] = __builtin_amdgcn_wmma_f32_16x16x32_f16(
                        false, a[m], false, b[n], (short)0, c[m][n], false, false);
        }

        // ---- epilogue: exp + masks, accumulate into per-lane column registers ----
        #pragma unroll
        for (int m = 0; m < 4; ++m) {
            const int rbase = wm * 64 + m * 16 + 8 * hh;   // 8 consecutive rows
            i32x4 l0 = *(const i32x4*)(labI + rbase);
            i32x4 l1 = *(const i32x4*)(labI + rbase + 4);
            const int li[8] = { l0.x, l0.y, l0.z, l0.w, l1.x, l1.y, l1.z, l1.w };
            const int ibase = ib0 + rbase;
            #pragma unroll
            for (int n = 0; n < 4; ++n) {
                #pragma unroll
                for (int r = 0; r < 8; ++r) {
                    const float e = __expf(c[m][n][r]);
                    const bool same = (li[r] == lj[n]);
                    cD[n] += same ? 0.0f : e;
                    cS[n] += (same && (ibase + r != jg[n])) ? e : 0.0f;
                }
            }
        }
    }

    // ---- flush: combine 8 contributing lanes per column in LDS, then global ----
    __syncthreads();
    #pragma unroll
    for (int n = 0; n < 4; ++n) {
        const int jc = wn * 64 + n * 16 + l16;
        atomicAdd(&accSl[jc], cS[n]);
        atomicAdd(&accDl[jc], cD[n]);
    }
    __syncthreads();
    if (t < BLK) {
        atomicAdd(&row_same[j0 + t], accSl[t]);
        atomicAdd(&row_diff[j0 + t], accDl[t]);
    }
}

// ---------------- kernel 3: per-class segment sums + final scalar ----------------
__global__ __launch_bounds__(256)
void finalize_kernel(const float* __restrict__ row_same,
                     const float* __restrict__ row_diff,
                     const int* __restrict__ labels,
                     float* __restrict__ out) {
    __shared__ float segS[NUM_CLASSES];
    __shared__ float segD[NUM_CLASSES];
    __shared__ int   cnt[NUM_CLASSES];
    __shared__ float red[256];
    const int t = threadIdx.x;
    if (t < NUM_CLASSES) { segS[t] = 0.0f; segD[t] = 0.0f; cnt[t] = 0; }
    __syncthreads();
    for (int r = t; r < N_ROWS; r += 256) {
        const int lab = labels[r];
        atomicAdd(&segS[lab], row_same[r]);
        atomicAdd(&segD[lab], row_diff[r]);
        atomicAdd(&cnt[lab], 1);
    }
    __syncthreads();
    float v = 0.0f;
    if (t < NUM_CLASSES && cnt[t] > 0) v = __logf(segS[t] / segD[t]);
    red[t] = v;
    __syncthreads();
    #pragma unroll
    for (int s = 128; s >= 1; s >>= 1) {
        if (t < s) red[t] += red[t + s];
        __syncthreads();
    }
    if (t == 0) out[0] = red[0] / (float)NUM_BATCHES;
}

// ---------------- launch ----------------
extern "C" void kernel_launch(void* const* d_in, const int* in_sizes, int n_in,
                              void* d_out, int out_size, void* d_ws, size_t ws_size,
                              hipStream_t stream) {
    const float* z      = (const float*)d_in[1];   // semantic_embeddings [8192,1024] f32
    const int*   labels = (const int*)d_in[3];     // labels [8192]
    float* out = (float*)d_out;

    _Float16* zn = (_Float16*)d_ws;                              // 16 MB
    float* row_same = (float*)((char*)d_ws + (size_t)N_ROWS * DIM * sizeof(_Float16));
    float* row_diff = row_same + N_ROWS;

    zero_acc_kernel<<<(2 * N_ROWS + 255) / 256, 256, 0, stream>>>(row_same, 2 * N_ROWS);
    normalize_rows_kernel<<<N_ROWS, 256, 0, stream>>>(z, zn);
    gram_epilogue_kernel<<<dim3(N_ROWS / BLK, ISPLIT), NTHREADS, 0, stream>>>(
        zn, labels, row_same, row_diff);
    finalize_kernel<<<1, 256, 0, stream>>>(row_same, row_diff, labels, out);
}
